// VAE_21715354649531
// MI455X (gfx1250) — matrix operand order repaired and checked by reading, not compiled, not run on hardware
//
#include <hip/hip_runtime.h>

// ============================================================================
// VAE seq2seq (bi-LSTM encoders + LSTMCell decoder) optimized for MI455X
// (gfx1250, wave32, WMMA).
//
// Reasoning (no runtime available -- from MI455X specs):
//  * ~1 TFLOP of GEMM work; weights ~17 MB in bf16 => resident in 192 MB L2
//    for all 428 recurrence steps => per-step GEMMs are matrix-math bound.
//  * bf16 WMMA (v_wmma_f32_16x16x32_bf16, f32 accumulate) is the fast path;
//    fp32 VALU or the K=4 f32 WMMA would be far slower.
//  * Recurrent dependency h_t -> h_{t+1} forces per-step kernel launches;
//    each GEMM: 64x64 C tile per 128-thread WG, 2x2 wave grid, 32x32
//    accumulator per wave = 4 v_wmma tiles per K-chunk of 32.
//  * Decoder optimization: split dr_wih columns; the concat(h_x,z) part
//    (K=640) is computed once into Gconst; per step only K=64 (y_prev) and
//    K=512 (h) gate GEMMs remain.
// ============================================================================

typedef __attribute__((ext_vector_type(16))) __bf16 v16bf;
typedef __attribute__((ext_vector_type(8)))  __bf16 v8bf;
typedef __attribute__((ext_vector_type(8)))  float  v8f;

#define NXD 64
#define NYD 64
#define NZD 128
#define NHD 512
#define HBD 256     // per-direction encoder hidden
#define TXS 64
#define TYS 100
#define BATCH 1024
#define HOR 100
#define MH1D 1024
#define MH2D 512

// ---------------------------------------------------------------------------
// Device helpers
// ---------------------------------------------------------------------------
__device__ __forceinline__ float sigmoidf_(float x) {
  return 1.0f / (1.0f + __expf(-x));
}

// A-operand fragment: 16x32 bf16 tile at (row0, k0) of row-major A[.., lda].
// CDNA5 layout (ISA 7.12.2, 16-bit A 16x32): lane m = lane&15, half = lane>>4,
// element j: K = (j<8 ? j : j+8) + half*8.
__device__ __forceinline__ v16bf load_A_frag(const __bf16* __restrict__ A,
                                             int lda, int row0, int k0,
                                             int lane) {
  const int m = row0 + (lane & 15);
  const int half = lane >> 4;
  const __bf16* p = A + (size_t)m * lda + k0 + half * 8;
  v8bf lo = *reinterpret_cast<const v8bf*>(p);       // K = k0 + half*8 + 0..7
  v8bf hi = *reinterpret_cast<const v8bf*>(p + 16);  // K = k0+16+half*8 + 0..7
  return __builtin_shufflevector(lo, hi, 0, 1, 2, 3, 4, 5, 6, 7, 8, 9, 10, 11,
                                 12, 13, 14, 15);
}

// B-operand fragment: 32x16 (KxN) from torch-style W[N, ldw] row-major, so
// B[k][n] = W[n][k] and each lane reads its output-column's weight row
// contiguously in K. Layout (matches A): lane n = lane&15 holds
// K = (j<8 ? j : j+8) + half*8.
__device__ __forceinline__ v16bf load_B_frag(const __bf16* __restrict__ W,
                                             int ldw, int col0, int k0,
                                             int lane) {
  const int n = col0 + (lane & 15);
  const int half = lane >> 4;
  const __bf16* p = W + (size_t)n * ldw + k0 + half * 8;
  v8bf lo = *reinterpret_cast<const v8bf*>(p);
  v8bf hi = *reinterpret_cast<const v8bf*>(p + 16);
  return __builtin_shufflevector(lo, hi, 0, 1, 2, 3, 4, 5, 6, 7, 8, 9, 10, 11,
                                 12, 13, 14, 15);
}

// ---------------------------------------------------------------------------
// Tiled WMMA GEMM:  C[M,N](f32) (+=/=) A[M,K](bf16) @ W[N,K]^T(bf16) + bias
// Optionally fused tanh; optionally mirrors result to a bf16 buffer so the
// next GEMM in the chain can consume it without a conversion pass.
// Requires M%64==0, N%64==0, K%32==0 (true for every GEMM in this model).
// ---------------------------------------------------------------------------
__global__ __launch_bounds__(128) void wmma_gemm64_kernel(
    const __bf16* __restrict__ A, int lda, const __bf16* __restrict__ W,
    int ldw, float* __restrict__ C, int ldc, __bf16* __restrict__ Cbf,
    int ldcbf, const float* __restrict__ bias, int K, int accumulate,
    int fuse_tanh) {
  const int lane = threadIdx.x & 31;
  const int wave = threadIdx.x >> 5;
  const int m0 = blockIdx.y * 64 + (wave >> 1) * 32;
  const int n0 = blockIdx.x * 64 + (wave & 1) * 32;

  v8f acc00 = {}, acc01 = {}, acc10 = {}, acc11 = {};

  for (int k0 = 0; k0 < K; k0 += 32) {
    v16bf a0 = load_A_frag(A, lda, m0, k0, lane);
    v16bf a1 = load_A_frag(A, lda, m0 + 16, k0, lane);
    v16bf b0 = load_B_frag(W, ldw, n0, k0, lane);
    v16bf b1 = load_B_frag(W, ldw, n0 + 16, k0, lane);
    acc00 = __builtin_amdgcn_wmma_f32_16x16x32_bf16(false, a0, false, b0,
                                                    (short)0, acc00, false,
                                                    false);
    acc01 = __builtin_amdgcn_wmma_f32_16x16x32_bf16(false, a0, false, b1,
                                                    (short)0, acc01, false,
                                                    false);
    acc10 = __builtin_amdgcn_wmma_f32_16x16x32_bf16(false, a1, false, b0,
                                                    (short)0, acc10, false,
                                                    false);
    acc11 = __builtin_amdgcn_wmma_f32_16x16x32_bf16(false, a1, false, b1,
                                                    (short)0, acc11, false,
                                                    false);
  }

  // C/D layout: lane n = lane&15; VGPR r -> m = r + 8*(lane>>4).
  const int nn = lane & 15;
  const int half = lane >> 4;
#pragma unroll
  for (int mi = 0; mi < 2; ++mi) {
#pragma unroll
    for (int ni = 0; ni < 2; ++ni) {
      v8f accv = (mi == 0) ? (ni == 0 ? acc00 : acc01)
                           : (ni == 0 ? acc10 : acc11);
#pragma unroll
      for (int r = 0; r < 8; ++r) {
        const int row = m0 + mi * 16 + r + half * 8;
        const int col = n0 + ni * 16 + nn;
        float v = accv[r];
        if (bias) v += bias[col];
        if (accumulate) v += C[(size_t)row * ldc + col];
        if (fuse_tanh) v = tanhf(v);
        if (C) C[(size_t)row * ldc + col] = v;
        if (Cbf) Cbf[(size_t)row * ldcbf + col] = (__bf16)v;
      }
    }
  }
}

// ---------------------------------------------------------------------------
// Pointwise LSTM cell: g (+ optional gconst) + bih + bhh -> update c, emit
// bf16 h (for the next WMMA GEMM) and optionally accumulate h into a running
// time-mean buffer (encoder).  Gate order i, f, g, o (torch LSTMCell).
// ---------------------------------------------------------------------------
__global__ void lstm_cell_kernel(const float* __restrict__ g,
                                 const float* __restrict__ gconst,
                                 const float* __restrict__ bih,
                                 const float* __restrict__ bhh,
                                 float* __restrict__ c,
                                 __bf16* __restrict__ h_bf,
                                 float* __restrict__ hsum, int sum_ld,
                                 int sum_off, int H, int n) {
  const int idx = blockIdx.x * blockDim.x + threadIdx.x;
  if (idx >= n) return;
  const int b = idx / H;
  const int j = idx - b * H;
  const float* gr = g + (size_t)b * 4 * H;
  float gi = gr[j] + bih[j] + bhh[j];
  float gf = gr[H + j] + bih[H + j] + bhh[H + j];
  float gg = gr[2 * H + j] + bih[2 * H + j] + bhh[2 * H + j];
  float go = gr[3 * H + j] + bih[3 * H + j] + bhh[3 * H + j];
  if (gconst) {
    const float* gc = gconst + (size_t)b * 4 * H;
    gi += gc[j];
    gf += gc[H + j];
    gg += gc[2 * H + j];
    go += gc[3 * H + j];
  }
  float cc = c[idx];
  cc = sigmoidf_(gf) * cc + sigmoidf_(gi) * tanhf(gg);
  const float hh = sigmoidf_(go) * tanhf(cc);
  c[idx] = cc;
  h_bf[idx] = (__bf16)hh;
  if (hsum) hsum[(size_t)b * sum_ld + sum_off + j] += hh;
}

// ---------------------------------------------------------------------------
// Small utility kernels
// ---------------------------------------------------------------------------
__global__ void cvt_f32_bf16_kernel(__bf16* __restrict__ dst,
                                    const float* __restrict__ src, int n) {
  const int i = blockIdx.x * blockDim.x + threadIdx.x;
  if (i < n) dst[i] = (__bf16)src[i];
}

__global__ void zero_f32_kernel(float* __restrict__ p, int n) {
  const int i = blockIdx.x * blockDim.x + threadIdx.x;
  if (i < n) p[i] = 0.0f;
}

__global__ void zero_bf16_kernel(__bf16* __restrict__ p, int n) {
  const int i = blockIdx.x * blockDim.x + threadIdx.x;
  if (i < n) p[i] = (__bf16)0.0f;
}

// henc[b, 0:NH)  = hx_sum * invTx   (h_x, fwd||bwd halves already concatenated)
// henc[b, NH:2NH)= hy_sum * invTy   (h_y)
__global__ void finalize_means_kernel(__bf16* __restrict__ henc,
                                      const float* __restrict__ hx,
                                      const float* __restrict__ hy,
                                      float invTx, float invTy, int n) {
  const int i = blockIdx.x * blockDim.x + threadIdx.x;
  if (i >= n) return;
  const int b = i / (2 * NHD);
  const int j = i - b * (2 * NHD);
  const float v = (j < NHD) ? hx[(size_t)b * NHD + j] * invTx
                            : hy[(size_t)b * NHD + (j - NHD)] * invTy;
  henc[i] = (__bf16)v;
}

// dconst[b] = concat(h_x[b] (512, bf16 from henc cols 0..511), z[b] (128))
__global__ void build_dconst_kernel(__bf16* __restrict__ dc,
                                    const __bf16* __restrict__ henc,
                                    const __bf16* __restrict__ zbf, int n) {
  const int i = blockIdx.x * blockDim.x + threadIdx.x;
  if (i >= n) return;
  const int b = i / (NHD + NZD);
  const int j = i - b * (NHD + NZD);
  dc[i] = (j < NHD) ? henc[(size_t)b * 2 * NHD + j]
                    : zbf[(size_t)b * NZD + (j - NHD)];
}

// ---------------------------------------------------------------------------
// Host orchestration
// ---------------------------------------------------------------------------
extern "C" void kernel_launch(void* const* d_in, const int* in_sizes, int n_in,
                              void* d_out, int out_size, void* d_ws,
                              size_t ws_size, hipStream_t stream) {
  (void)in_sizes;
  (void)n_in;
  (void)out_size;

  const float* x = (const float*)d_in[0];
  const float* y = (const float*)d_in[1];
  const float* xr_wih_f = (const float*)d_in[2];
  const float* xr_whh_f = (const float*)d_in[3];
  const float* xr_bih_f = (const float*)d_in[4];
  const float* xr_bhh_f = (const float*)d_in[5];
  const float* xr_wih_b = (const float*)d_in[6];
  const float* xr_whh_b = (const float*)d_in[7];
  const float* xr_bih_b = (const float*)d_in[8];
  const float* xr_bhh_b = (const float*)d_in[9];
  const float* er_wih_f = (const float*)d_in[10];
  const float* er_whh_f = (const float*)d_in[11];
  const float* er_bih_f = (const float*)d_in[12];
  const float* er_bhh_f = (const float*)d_in[13];
  const float* er_wih_b = (const float*)d_in[14];
  const float* er_whh_b = (const float*)d_in[15];
  const float* er_bih_b = (const float*)d_in[16];
  const float* er_bhh_b = (const float*)d_in[17];
  const float* emlp_w1 = (const float*)d_in[18];
  const float* emlp_b1 = (const float*)d_in[19];
  const float* emlp_w2 = (const float*)d_in[20];
  const float* emlp_b2 = (const float*)d_in[21];
  const float* emu_w = (const float*)d_in[22];
  const float* emu_b = (const float*)d_in[23];
  const float* elv_w = (const float*)d_in[24];
  const float* elv_b = (const float*)d_in[25];
  const float* dr_wih = (const float*)d_in[26];
  const float* dr_whh = (const float*)d_in[27];
  const float* dr_bih = (const float*)d_in[28];
  const float* dr_bhh = (const float*)d_in[29];
  const float* dmlp_w1 = (const float*)d_in[30];
  const float* dmlp_b1 = (const float*)d_in[31];
  const float* dmlp_w2 = (const float*)d_in[32];
  const float* dmlp_b2 = (const float*)d_in[33];
  const float* dout_w = (const float*)d_in[34];
  const float* dout_b = (const float*)d_in[35];

  float* out = (float*)d_out;
  float* y_out = out;                                   // [100, 1024, 64]
  float* mu_out = out + (size_t)HOR * BATCH * NYD;      // [1024, 128]
  float* lv_out = mu_out + (size_t)BATCH * NZD;         // [1024, 128]

  // ---- workspace arena (pointer math only; 256-B aligned slices) ----
  char* wsb = (char*)d_ws;
  size_t wsoff = 0;
  auto alloc = [&](size_t bytes) -> void* {
    wsoff = (wsoff + 255) & ~(size_t)255;
    void* p = wsb + wsoff;
    wsoff += bytes;
    return p;
  };
  auto abf = [&](size_t n) { return (__bf16*)alloc(n * sizeof(__bf16)); };
  auto af32 = [&](size_t n) { return (float*)alloc(n * sizeof(float)); };

  // bf16 mirrors of sequences + weights (built per launch; deterministic)
  __bf16* x_bf = abf((size_t)TXS * BATCH * NXD);
  __bf16* y_bf = abf((size_t)TYS * BATCH * NYD);
  __bf16* xr_wih_f_bf = abf((size_t)4 * HBD * NXD);
  __bf16* xr_whh_f_bf = abf((size_t)4 * HBD * HBD);
  __bf16* xr_wih_b_bf = abf((size_t)4 * HBD * NXD);
  __bf16* xr_whh_b_bf = abf((size_t)4 * HBD * HBD);
  __bf16* er_wih_f_bf = abf((size_t)4 * HBD * NYD);
  __bf16* er_whh_f_bf = abf((size_t)4 * HBD * HBD);
  __bf16* er_wih_b_bf = abf((size_t)4 * HBD * NYD);
  __bf16* er_whh_b_bf = abf((size_t)4 * HBD * HBD);
  __bf16* emlp_w1_bf = abf((size_t)MH1D * 2 * NHD);
  __bf16* emlp_w2_bf = abf((size_t)MH2D * MH1D);
  __bf16* emu_w_bf = abf((size_t)NZD * MH2D);
  __bf16* elv_w_bf = abf((size_t)NZD * MH2D);
  __bf16* dr_wih_bf = abf((size_t)4 * NHD * (NYD + NZD + NHD));
  __bf16* dr_whh_bf = abf((size_t)4 * NHD * NHD);
  __bf16* dmlp_w1_bf = abf((size_t)MH1D * NHD);
  __bf16* dmlp_w2_bf = abf((size_t)MH2D * MH1D);
  __bf16* dout_w_bf = abf((size_t)NYD * MH2D);

  // activations / state
  float* Gbuf = af32((size_t)BATCH * 4 * NHD);    // gate buffer (max 4*512)
  float* Gconst = af32((size_t)BATCH * 4 * NHD);  // decoder constant gates
  float* hx_sum = af32((size_t)BATCH * NHD);
  float* hy_sum = af32((size_t)BATCH * NHD);
  float* cbuf = af32((size_t)BATCH * NHD);
  __bf16* h_bf = abf((size_t)BATCH * NHD);
  __bf16* henc_bf = abf((size_t)BATCH * 2 * NHD);
  __bf16* m1_bf = abf((size_t)BATCH * MH1D);
  __bf16* m2_bf = abf((size_t)BATCH * MH2D);
  __bf16* z_bf = abf((size_t)BATCH * NZD);
  __bf16* dconst_bf = abf((size_t)BATCH * (NHD + NZD));
  __bf16* yp_bf = abf((size_t)BATCH * NYD);

  if (wsoff > ws_size) return;  // not enough scratch; bail deterministically

  // ---- launch helpers ----
  auto cvt = [&](__bf16* dst, const float* src, size_t n) {
    cvt_f32_bf16_kernel<<<dim3((unsigned)((n + 255) / 256)), dim3(256), 0,
                          stream>>>(dst, src, (int)n);
  };
  auto zf32 = [&](float* p, size_t n) {
    zero_f32_kernel<<<dim3((unsigned)((n + 255) / 256)), dim3(256), 0,
                      stream>>>(p, (int)n);
  };
  auto zbf = [&](__bf16* p, size_t n) {
    zero_bf16_kernel<<<dim3((unsigned)((n + 255) / 256)), dim3(256), 0,
                       stream>>>(p, (int)n);
  };
  auto gemm = [&](const __bf16* A, int lda, const __bf16* W, int ldw, float* C,
                  int ldc, __bf16* Cbf, int ldcbf, const float* bias, int M,
                  int N, int K, int accF, int tanhF) {
    wmma_gemm64_kernel<<<dim3(N / 64, M / 64), dim3(128), 0, stream>>>(
        A, lda, W, ldw, C, ldc, Cbf, ldcbf, bias, K, accF, tanhF);
  };
  auto lstm = [&](const float* g, const float* gc, const float* bih,
                  const float* bhh, float* c, __bf16* hb, float* hsum,
                  int sum_ld, int sum_off, int H) {
    const int n = BATCH * H;
    lstm_cell_kernel<<<dim3((n + 255) / 256), dim3(256), 0, stream>>>(
        g, gc, bih, bhh, c, hb, hsum, sum_ld, sum_off, H, n);
  };

  // ---- 1) fp32 -> bf16 conversions ----
  cvt(x_bf, x, (size_t)TXS * BATCH * NXD);
  cvt(y_bf, y, (size_t)TYS * BATCH * NYD);
  cvt(xr_wih_f_bf, xr_wih_f, (size_t)4 * HBD * NXD);
  cvt(xr_whh_f_bf, xr_whh_f, (size_t)4 * HBD * HBD);
  cvt(xr_wih_b_bf, xr_wih_b, (size_t)4 * HBD * NXD);
  cvt(xr_whh_b_bf, xr_whh_b, (size_t)4 * HBD * HBD);
  cvt(er_wih_f_bf, er_wih_f, (size_t)4 * HBD * NYD);
  cvt(er_whh_f_bf, er_whh_f, (size_t)4 * HBD * HBD);
  cvt(er_wih_b_bf, er_wih_b, (size_t)4 * HBD * NYD);
  cvt(er_whh_b_bf, er_whh_b, (size_t)4 * HBD * HBD);
  cvt(emlp_w1_bf, emlp_w1, (size_t)MH1D * 2 * NHD);
  cvt(emlp_w2_bf, emlp_w2, (size_t)MH2D * MH1D);
  cvt(emu_w_bf, emu_w, (size_t)NZD * MH2D);
  cvt(elv_w_bf, elv_w, (size_t)NZD * MH2D);
  cvt(dr_wih_bf, dr_wih, (size_t)4 * NHD * (NYD + NZD + NHD));
  cvt(dr_whh_bf, dr_whh, (size_t)4 * NHD * NHD);
  cvt(dmlp_w1_bf, dmlp_w1, (size_t)MH1D * NHD);
  cvt(dmlp_w2_bf, dmlp_w2, (size_t)MH2D * MH1D);
  cvt(dout_w_bf, dout_w, (size_t)NYD * MH2D);

  // ---- 2) bi-LSTM encoders (per-step: 2 WMMA GEMMs + pointwise cell) ----
  zf32(hx_sum, (size_t)BATCH * NHD);
  zf32(hy_sum, (size_t)BATCH * NHD);

  auto run_dir = [&](const __bf16* seq, int T, int in_dim, const __bf16* wih,
                     const __bf16* whh, const float* bih, const float* bhh,
                     float* hsum, int sum_off, bool rev) {
    zbf(h_bf, (size_t)BATCH * HBD);
    zf32(cbuf, (size_t)BATCH * HBD);
    for (int s = 0; s < T; ++s) {
      const int t = rev ? (T - 1 - s) : s;
      // gates = x_t @ wih^T  (overwrite)
      gemm(seq + (size_t)t * BATCH * in_dim, in_dim, wih, in_dim, Gbuf,
           4 * HBD, nullptr, 0, nullptr, BATCH, 4 * HBD, in_dim, 0, 0);
      // gates += h @ whh^T  (accumulate)
      gemm(h_bf, HBD, whh, HBD, Gbuf, 4 * HBD, nullptr, 0, nullptr, BATCH,
           4 * HBD, HBD, 1, 0);
      lstm(Gbuf, nullptr, bih, bhh, cbuf, h_bf, hsum, NHD, sum_off, HBD);
    }
  };

  run_dir(x_bf, TXS, NXD, xr_wih_f_bf, xr_whh_f_bf, xr_bih_f, xr_bhh_f, hx_sum,
          0, false);
  run_dir(x_bf, TXS, NXD, xr_wih_b_bf, xr_whh_b_bf, xr_bih_b, xr_bhh_b, hx_sum,
          HBD, true);
  run_dir(y_bf, TYS, NYD, er_wih_f_bf, er_whh_f_bf, er_bih_f, er_bhh_f, hy_sum,
          0, false);
  run_dir(y_bf, TYS, NYD, er_wih_b_bf, er_whh_b_bf, er_bih_b, er_bhh_b, hy_sum,
          HBD, true);

  {
    const int n = BATCH * 2 * NHD;
    finalize_means_kernel<<<dim3((n + 255) / 256), dim3(256), 0, stream>>>(
        henc_bf, hx_sum, hy_sum, 1.0f / (float)TXS, 1.0f / (float)TYS, n);
  }

  // ---- 3) encoder MLP + mu / logvar heads (tanh fused in GEMM epilogue) ----
  gemm(henc_bf, 2 * NHD, emlp_w1_bf, 2 * NHD, nullptr, 0, m1_bf, MH1D, emlp_b1,
       BATCH, MH1D, 2 * NHD, 0, 1);
  gemm(m1_bf, MH1D, emlp_w2_bf, MH1D, nullptr, 0, m2_bf, MH2D, emlp_b2, BATCH,
       MH2D, MH1D, 0, 1);
  gemm(m2_bf, MH2D, emu_w_bf, MH2D, mu_out, NZD, z_bf, NZD, emu_b, BATCH, NZD,
       MH2D, 0, 0);  // z = mu (eval mode), also emitted as bf16
  gemm(m2_bf, MH2D, elv_w_bf, MH2D, lv_out, NZD, nullptr, 0, elv_b, BATCH, NZD,
       MH2D, 0, 0);

  // ---- 4) decoder: hoist constant part of the input projection ----
  {
    const int n = BATCH * (NHD + NZD);
    build_dconst_kernel<<<dim3((n + 255) / 256), dim3(256), 0, stream>>>(
        dconst_bf, henc_bf, z_bf, n);
  }
  // Gconst = concat(h_x, z) @ dr_wih[:, 0:640]^T     (once, K=640)
  gemm(dconst_bf, NHD + NZD, dr_wih_bf, NYD + NZD + NHD, Gconst, 4 * NHD,
       nullptr, 0, nullptr, BATCH, 4 * NHD, NHD + NZD, 0, 0);

  zbf(h_bf, (size_t)BATCH * NHD);
  zf32(cbuf, (size_t)BATCH * NHD);
  cvt(yp_bf, x + (size_t)(TXS - 1) * BATCH * NXD, (size_t)BATCH * NXD);

  const __bf16* dr_wy_bf = dr_wih_bf + (NHD + NZD);  // cols 640..703 (y_prev)

  for (int t = 0; t < HOR; ++t) {
    // gates = yp @ Wy^T (K=64, overwrite) ; gates += h @ dr_whh^T (K=512)
    gemm(yp_bf, NYD, dr_wy_bf, NYD + NZD + NHD, Gbuf, 4 * NHD, nullptr, 0,
         nullptr, BATCH, 4 * NHD, NYD, 0, 0);
    gemm(h_bf, NHD, dr_whh_bf, NHD, Gbuf, 4 * NHD, nullptr, 0, nullptr, BATCH,
         4 * NHD, NHD, 1, 0);
    lstm(Gbuf, Gconst, dr_bih, dr_bhh, cbuf, h_bf, nullptr, 0, 0, NHD);
    // decoder MLP (tanh fused) + output head; head writes f32 to d_out and
    // bf16 back into yp for the next step's gate GEMM.
    gemm(h_bf, NHD, dmlp_w1_bf, NHD, nullptr, 0, m1_bf, MH1D, dmlp_b1, BATCH,
         MH1D, NHD, 0, 1);
    gemm(m1_bf, MH1D, dmlp_w2_bf, MH1D, nullptr, 0, m2_bf, MH2D, dmlp_b2,
         BATCH, MH2D, MH1D, 0, 1);
    gemm(m2_bf, MH2D, dout_w_bf, MH2D, y_out + (size_t)t * BATCH * NYD, NYD,
         yp_bf, NYD, dout_b, BATCH, NYD, MH2D, 0, 0);
  }
}
